// CentroidEstimationModel_3977139716683
// MI455X (gfx1250) — compile-verified, matrix operand order checked by reading, not run on hardware
//
#include <hip/hip_runtime.h>
#include <hip/hip_bf16.h>
#include <stdint.h>

// ---------------------------------------------------------------------------
// CDNA5 (gfx1250) wave32 WMMA types
// ---------------------------------------------------------------------------
typedef __attribute__((ext_vector_type(16))) __bf16 v16bf;
typedef __attribute__((ext_vector_type(8)))  float  v8f;

union Frag {
    unsigned u[8];
    uint4    q[2];
    v16bf    v;
};

__device__ inline unsigned short f2bf(float f) {
    unsigned u = __float_as_uint(f);
    u += 0x7FFFu + ((u >> 16) & 1u);          // round-to-nearest-even
    return (unsigned short)(u >> 16);
}
__device__ inline float bf2f(unsigned short h) {
    return __uint_as_float(((unsigned)h) << 16);
}
// wave32 all-reduce sum
__device__ inline float wred(float v) {
#pragma unroll
    for (int m = 16; m >= 1; m >>= 1) v += __shfl_xor(v, m, 32);
    return v;
}

// ---------------------------------------------------------------------------
// Prep: convert f32 weights -> bf16 pre-swizzled into WMMA B-fragment layout.
// B (32x16 per k-tile): lane%16 = N col, lane/16 selects K half, VGPR p holds
// K = (p<4 ? 2p : 2p+8) + 8*(lane/16) + {0,1}.  Stored so each fragment is a
// contiguous 32-byte chunk: wsw[((kt*32+nt)*32+lane)*16 + e].
// ---------------------------------------------------------------------------
__global__ __launch_bounds__(256) void prep_w_kernel(
    const float* __restrict__ w, unsigned short* __restrict__ wsw,
    int ktiles, int krows)
{
    int gid = blockIdx.x * 256 + threadIdx.x;
    int total = ktiles * 32 * 32;
    if (gid >= total) return;
    int lane = gid & 31, nt = (gid >> 5) & 31, kt = gid >> 10;
    int half = lane >> 4, col = nt * 16 + (lane & 15);
    Frag fr;
    unsigned short* t = (unsigned short*)fr.u;
#pragma unroll
    for (int e = 0; e < 16; ++e) {
        int vv = e >> 1;
        int k = kt * 32 + ((vv < 4) ? 2 * vv : 2 * vv + 8) + 8 * half + (e & 1);
        float val = (k < krows) ? w[k * 512 + col] : 0.f;
        t[e] = f2bf(val);
    }
    uint4* dst = (uint4*)(wsw + (size_t)gid * 16);
    dst[0] = fr.q[0]; dst[1] = fr.q[1];
}

// ---------------------------------------------------------------------------
// Main per-batch kernel: one workgroup (8 wave32) per batch b.
// ---------------------------------------------------------------------------
__global__ __launch_bounds__(256) void centroid_main_kernel(
    const float* __restrict__ x, const unsigned char* __restrict__ amask,
    const int* __restrict__ order, const float* __restrict__ num_docs,
    const float* __restrict__ dweights, const float* __restrict__ pos_emb,
    const float* __restrict__ fc1_b1, const float* __restrict__ fc1_w2,
    const float* __restrict__ ln1_g, const float* __restrict__ ln1_b,
    const float* __restrict__ ln2_g, const float* __restrict__ ln2_b,
    const unsigned short* __restrict__ w1sw,
    unsigned short* __restrict__ asw, float* __restrict__ headmean)
{
    constexpr int N = 30, Dd = 512, KP = 1056;          // K padded 1025->1056
    extern __shared__ unsigned char smem[];
    float*          x1s  = (float*)smem;                       // [30][512] f32
    unsigned short* Abuf = (unsigned short*)(smem + 61440);    // [32][1056] bf16
    float*          mp   = (float*)(smem + 129024);            // [512]
    float*          mpx  = (float*)(smem + 131072);            // [512]
    float*          xpn2 = (float*)(smem + 133120);            // [32]
    float*          w2s  = (float*)(smem + 133248);            // [512][4]
    float*          Zs   = (float*)(smem + 141440);            // [32][4]
    float*          Asm  = (float*)(smem + 141952);            // [32][4]
    float*          Hh   = (float*)(smem + 142464);            // [2048]
    float*          red  = (float*)(smem + 150656);            // [32]

    int tid = threadIdx.x, w = tid >> 5, lane = tid & 31;
    int b = blockIdx.x;

    // ---- init LDS ----
    unsigned* az = (unsigned*)Abuf;
    for (int i = tid; i < 32 * KP / 2; i += 256) az[i] = 0u;
    for (int i = tid; i < Dd; i += 256) { mp[i] = 0.f; mpx[i] = 0.f; }
    for (int i = tid; i < 2048; i += 256) w2s[i] = fc1_w2[i];
    for (int i = tid; i < 128; i += 256) Zs[i] = 0.f;
    __syncthreads();

    // ---- phase 1: normalize -> LN1 -> +pos_emb; pools ----
    int c0 = lane * 16;
    float accmp[16], accmpx[16];
#pragma unroll
    for (int i = 0; i < 16; ++i) { accmp[i] = 0.f; accmpx[i] = 0.f; }
    const float* xb = x + (size_t)b * N * Dd;
    for (int n = w; n < N; n += 8) {
        float v[16];
#pragma unroll
        for (int i = 0; i < 16; ++i) v[i] = xb[n * Dd + c0 + i];
        float s2 = 0.f;
#pragma unroll
        for (int i = 0; i < 16; ++i) s2 += v[i] * v[i];
        s2 = wred(s2);
        float inv = 1.f / fmaxf(sqrtf(s2), 1e-8f);
        float s = 0.f;
#pragma unroll
        for (int i = 0; i < 16; ++i) { v[i] *= inv; s += v[i]; }
        float mu = wred(s) * (1.f / Dd);
        float sv = 0.f;
#pragma unroll
        for (int i = 0; i < 16; ++i) { float d = v[i] - mu; sv += d * d; }
        float rs = rsqrtf(wred(sv) * (1.f / Dd) + 1e-5f);
        float wgt = dweights[b * N + n];
        const float* pe = pos_emb + order[b * N + n] * Dd;
        float sxp = 0.f;
#pragma unroll
        for (int i = 0; i < 16; ++i) {
            float x1v = ln1_g[c0 + i] * (v[i] - mu) * rs + ln1_b[c0 + i];
            x1s[n * Dd + c0 + i] = x1v;
            accmpx[i] += wgt * x1v;
            float xp = x1v + pe[c0 + i];
            accmp[i] += wgt * xp;
            sxp += xp * xp;
            Abuf[n * KP + c0 + i] = f2bf(xp);
        }
        sxp = wred(sxp);
        if (lane == 0) xpn2[n] = sxp;
    }
#pragma unroll
    for (int i = 0; i < 16; ++i) {
        atomicAdd(&mp[c0 + i], accmp[i]);
        atomicAdd(&mpx[c0 + i], accmpx[i]);
    }
    __syncthreads();

    // ---- phase 2: finish pools, cosine feature, assemble concat input ----
    float rnd = 1.f / num_docs[b];
    for (int i = tid; i < Dd; i += 256) { mp[i] *= rnd; mpx[i] *= rnd; }
    __syncthreads();
    if (w == 0) {
        float pm = 0.f;
#pragma unroll
        for (int i = 0; i < 16; ++i) pm += mp[c0 + i] * mp[c0 + i];
        pm = wred(pm);
        if (lane == 0) red[24] = sqrtf(pm);           // ||mp||
    }
    for (int i = tid; i < N * Dd; i += 256) {
        int r = i >> 9, c = i & 511;
        Abuf[r * KP + Dd + c] = f2bf(mp[c]);          // broadcast mp cols 512..1023
    }
    __syncthreads();
    for (int n = w; n < N; n += 8) {
        float s = 0.f;
#pragma unroll
        for (int i = 0; i < 16; ++i) s += mp[c0 + i] * bf2f(Abuf[n * KP + c0 + i]);
        s = wred(s);
        if (lane == 0) {
            float den = fmaxf(red[24] * sqrtf(xpn2[n]), 1e-8f);
            Abuf[n * KP + 1024] = f2bf(s / den);      // cos column
        }
    }
    __syncthreads();

    // ---- phase 3: GEMM1  [32x1056]bf16 @ [1056x512]bf16 via v_wmma ----
    int half = lane >> 4, ln16 = lane & 15;
    v8f acc[2][4] = {};
    for (int kt = 0; kt < 33; ++kt) {
        Frag fa[2];
#pragma unroll
        for (int mt = 0; mt < 2; ++mt) {
            const unsigned* arow = (const unsigned*)(Abuf + (ln16 + 16 * mt) * KP);
#pragma unroll
            for (int p = 0; p < 8; ++p) {
                int kb = ((p < 4) ? 2 * p : 2 * p + 8) + 8 * half;   // ISA A-layout
                fa[mt].u[p] = arow[(kt * 32 + kb) >> 1];
            }
        }
#pragma unroll
        for (int j = 0; j < 4; ++j) {
            int nt = 4 * w + j;
            Frag fb;
            const uint4* pb = (const uint4*)(w1sw + ((size_t)(kt * 32 + nt) * 32 + lane) * 16);
            fb.q[0] = pb[0]; fb.q[1] = pb[1];
            acc[0][j] = __builtin_amdgcn_wmma_f32_16x16x32_bf16(
                false, fa[0].v, false, fb.v, (short)0, acc[0][j], false, false);
            acc[1][j] = __builtin_amdgcn_wmma_f32_16x16x32_bf16(
                false, fa[1].v, false, fb.v, (short)0, acc[1][j], false, false);
        }
    }
    // fused epilogue: tanh(+b1) then @ fc1_w2 (512x4) into Zs
#pragma unroll
    for (int mt = 0; mt < 2; ++mt)
#pragma unroll
        for (int j = 0; j < 4; ++j) {
            int col = (4 * w + j) * 16 + ln16;
            float b1 = fc1_b1[col];
#pragma unroll
            for (int r = 0; r < 8; ++r) {
                int m = mt * 16 + r + 8 * half;        // ISA C/D layout
                float t = tanhf(acc[mt][j][r] + b1);
#pragma unroll
                for (int h = 0; h < 4; ++h)
                    atomicAdd(&Zs[m * 4 + h], t * w2s[col * 4 + h]);
            }
        }
    __syncthreads();

    // ---- phase 4: softmax over N per head (fc1_b2 shift cancels) ----
    if (tid < 4) {
        int h = tid;
        float mx = -3.0e38f;
        for (int n = 0; n < N; ++n) {
            float zv = amask[b * N + n] ? -3.0e38f : Zs[n * 4 + h];
            Asm[n * 4 + h] = zv;
            mx = fmaxf(mx, zv);
        }
        float sum = 0.f;
        for (int n = 0; n < N; ++n) {
            float e = __expf(Asm[n * 4 + h] - mx);
            Asm[n * 4 + h] = e;
            sum += e;
        }
        float is = 1.f / sum;
        for (int n = 0; n < N; ++n) Asm[n * 4 + h] *= is;
    }
    __syncthreads();

    // ---- phase 5: Hh[h,d] = sum_n A_h[h,n]*x1[n,d] + mpx ----
    // reference reshape([B,N,H]->[B,H,N]) == flat index Asm[h*30+n]
#pragma unroll
    for (int i = 0; i < 8; ++i) {
        int o = tid + 256 * i;
        int h = o >> 9, d = o & 511;
        float a = 0.f;
        for (int n = 0; n < N; ++n) a += Asm[h * N + n] * x1s[n * Dd + d];
        Hh[o] = a + mpx[d];
    }
    __syncthreads();

    // ---- phase 6: LN2 over 2048, emit swizzled A-frags + head mean ----
    float s1 = 0.f, sq = 0.f;
#pragma unroll
    for (int i = 0; i < 8; ++i) { float v2 = Hh[tid + 256 * i]; s1 += v2; sq += v2 * v2; }
    s1 = wred(s1); sq = wred(sq);
    if (lane == 0) { red[w] = s1; red[8 + w] = sq; }
    __syncthreads();
    if (tid == 0) {
        float a0 = 0.f, a1 = 0.f;
        for (int i = 0; i < 8; ++i) { a0 += red[i]; a1 += red[8 + i]; }
        red[16] = a0; red[17] = a1;
    }
    __syncthreads();
    float mu2 = red[16] * (1.f / 2048.f);
    float var2 = red[17] * (1.f / 2048.f) - mu2 * mu2;
    float rs2 = rsqrtf(var2 + 1e-5f);
    int mtile = b >> 4, mrow = b & 15;
#pragma unroll
    for (int i = 0; i < 8; ++i) {
        int o = tid + 256 * i;
        float hn = ln2_g[o] * (Hh[o] - mu2) * rs2 + ln2_b[o];
        Hh[o] = hn;
        int kt = o >> 5, ko = o & 31;                  // inverse A-frag swizzle
        int hf, vv;
        if      (ko <  8) { hf = 0; vv = ko >> 1; }
        else if (ko < 16) { hf = 1; vv = (ko - 8) >> 1; }
        else if (ko < 24) { hf = 0; vv = 4 + ((ko - 16) >> 1); }
        else              { hf = 1; vv = 4 + ((ko - 24) >> 1); }
        int e = 2 * vv + (ko & 1);
        asw[((size_t)(mtile * 64 + kt) * 32 + (mrow + 16 * hf)) * 16 + e] = f2bf(hn);
    }
    __syncthreads();
    for (int d = tid; d < Dd; d += 256)
        headmean[(size_t)b * Dd + d] =
            0.25f * (Hh[d] + Hh[512 + d] + Hh[1024 + d] + Hh[1536 + d]);
}

// ---------------------------------------------------------------------------
// FC2: [4096x2048]bf16 @ [2048x512]bf16 via WMMA; fused LN3 -> +mean -> LN4.
// One block = one 16-row M-tile (16 batches).
// ---------------------------------------------------------------------------
__global__ __launch_bounds__(256) void fc2_kernel(
    const unsigned short* __restrict__ asw, const unsigned short* __restrict__ w2sw,
    const float* __restrict__ fc2_b, const float* __restrict__ headmean,
    const float* __restrict__ ln3_g, const float* __restrict__ ln3_b,
    const float* __restrict__ ln4_g, const float* __restrict__ ln4_b,
    float* __restrict__ out)
{
    __shared__ float preds[16 * 512];
    int tid = threadIdx.x, w = tid >> 5, lane = tid & 31;
    int mt = blockIdx.x;
    int half = lane >> 4, ln16 = lane & 15;
    v8f acc[4] = {};
    for (int kt = 0; kt < 64; ++kt) {
        Frag fa;
        const uint4* pa = (const uint4*)(asw + ((size_t)(mt * 64 + kt) * 32 + lane) * 16);
        fa.q[0] = pa[0]; fa.q[1] = pa[1];
#pragma unroll
        for (int j = 0; j < 4; ++j) {
            int nt = 4 * w + j;
            Frag fb;
            const uint4* pb = (const uint4*)(w2sw + ((size_t)(kt * 32 + nt) * 32 + lane) * 16);
            fb.q[0] = pb[0]; fb.q[1] = pb[1];
            acc[j] = __builtin_amdgcn_wmma_f32_16x16x32_bf16(
                false, fa.v, false, fb.v, (short)0, acc[j], false, false);
        }
    }
#pragma unroll
    for (int j = 0; j < 4; ++j) {
        int col = (4 * w + j) * 16 + ln16;
        float bb = fc2_b[col];
#pragma unroll
        for (int r = 0; r < 8; ++r) {
            int m = r + 8 * half;
            preds[m * 512 + col] = acc[j][r] + bb;
        }
    }
    __syncthreads();
#pragma unroll
    for (int rr = 0; rr < 2; ++rr) {
        int m = 2 * w + rr;
        int b = mt * 16 + m;
        int c0 = lane * 16;
        float v[16];
        float s = 0.f, sq = 0.f;
#pragma unroll
        for (int i = 0; i < 16; ++i) {
            v[i] = preds[m * 512 + c0 + i];
            s += v[i]; sq += v[i] * v[i];
        }
        s = wred(s); sq = wred(sq);
        float mu = s * (1.f / 512.f);
        float rs = rsqrtf(sq * (1.f / 512.f) - mu * mu + 1e-5f);
        float s2 = 0.f, sq2 = 0.f;
#pragma unroll
        for (int i = 0; i < 16; ++i) {
            v[i] = ln3_g[c0 + i] * (v[i] - mu) * rs + ln3_b[c0 + i]
                 + headmean[(size_t)b * 512 + c0 + i];
            s2 += v[i]; sq2 += v[i] * v[i];
        }
        s2 = wred(s2); sq2 = wred(sq2);
        float mu4 = s2 * (1.f / 512.f);
        float rs4 = rsqrtf(sq2 * (1.f / 512.f) - mu4 * mu4 + 1e-5f);
#pragma unroll
        for (int i = 0; i < 16; ++i)
            out[(size_t)b * 512 + c0 + i] =
                ln4_g[c0 + i] * (v[i] - mu4) * rs4 + ln4_b[c0 + i];
    }
}

// ---------------------------------------------------------------------------
extern "C" void kernel_launch(void* const* d_in, const int* in_sizes, int n_in,
                              void* d_out, int out_size, void* d_ws, size_t ws_size,
                              hipStream_t stream) {
    (void)in_sizes; (void)n_in; (void)out_size; (void)ws_size;
    const float*         x        = (const float*)d_in[0];
    const unsigned char* amask    = (const unsigned char*)d_in[1];
    const int*           order    = (const int*)d_in[2];
    const float*         num_docs = (const float*)d_in[3];
    const float*         dweights = (const float*)d_in[4];
    /* d_in[5] clusters_centroids: unused by reference */
    const float*         pos_emb  = (const float*)d_in[6];
    const float*         fc1_w1   = (const float*)d_in[7];
    const float*         fc1_b1   = (const float*)d_in[8];
    const float*         fc1_w2   = (const float*)d_in[9];
    /* d_in[10] fc1_b2: per-head constant, cancels in softmax over N */
    const float*         fc2_w    = (const float*)d_in[11];
    const float*         fc2_b    = (const float*)d_in[12];
    const float*         ln1_g    = (const float*)d_in[13];
    const float*         ln1_b    = (const float*)d_in[14];
    const float*         ln2_g    = (const float*)d_in[15];
    const float*         ln2_b    = (const float*)d_in[16];
    const float*         ln3_g    = (const float*)d_in[17];
    const float*         ln3_b    = (const float*)d_in[18];
    const float*         ln4_g    = (const float*)d_in[19];
    const float*         ln4_b    = (const float*)d_in[20];
    float* out = (float*)d_out;

    // workspace layout (bytes, 256-aligned); all regions fully rewritten per call
    unsigned char* ws = (unsigned char*)d_ws;
    unsigned short* w1sw = (unsigned short*)(ws + 0);          // 33*32*32*16 bf16 = 1,081,344 B
    unsigned short* w2sw = (unsigned short*)(ws + 1081344);    // 64*32*32*16 bf16 = 2,097,152 B
    unsigned short* asw  = (unsigned short*)(ws + 3178496);    // 256*64*32*16 bf16 = 16,777,216 B
    float*          hmean = (float*)(ws + 19955712);           // 4096*512 f32 = 8,388,608 B

    prep_w_kernel<<<132, 256, 0, stream>>>(fc1_w1, w1sw, 33, 1025);
    prep_w_kernel<<<256, 256, 0, stream>>>(fc2_w, w2sw, 64, 2048);

    constexpr unsigned SMEM_B = 150784;   // ~147 KB dynamic LDS (<= 320 KB/WGP)
    centroid_main_kernel<<<4096, 256, SMEM_B, stream>>>(
        x, amask, order, num_docs, dweights, pos_emb, fc1_b1, fc1_w2,
        ln1_g, ln1_b, ln2_g, ln2_b, w1sw, asw, hmean);

    fc2_kernel<<<256, 256, 0, stream>>>(asw, w2sw, fc2_b, hmean,
                                        ln3_g, ln3_b, ln4_g, ln4_b, out);
}
